// SelfAttention_12292196401708
// MI455X (gfx1250) — matrix operand order repaired and checked by reading, not compiled, not run on hardware
//
#include <hip/hip_runtime.h>

typedef _Float16 v16h __attribute__((ext_vector_type(16)));
typedef _Float16 v8h  __attribute__((ext_vector_type(8)));
typedef float    v8f  __attribute__((ext_vector_type(8)));
typedef float    v4f  __attribute__((ext_vector_type(4)));

#define B_   8
#define C_   256
#define D_   32
#define N_   4096
// 1/sqrt(32) * log2(e): Q pre-scale so scores are already in the exp2 domain
#define QSCALE_ (0.1767766952966369f * 1.4426950408889634f)

static __device__ __forceinline__ v8f wmma_f16(v16h a, v16h b, v8f c) {
    // D = A(16x32 f16) * B(32x16 f16) + C(16x16 f32)
    return __builtin_amdgcn_wmma_f32_16x16x32_f16(
        /*neg_a=*/false, a, /*neg_b=*/false, b,
        /*c_mod=*/(short)0, c, /*reuse_a=*/false, /*reuse_b=*/false);
}

// Load f16 A-operand (16x32, MxK) for one lane from a row-major [*, 32] row.
// ISA layout: lane L holds row M=L%16; h=L/16 selects K chunks {8h..8h+7, 16+8h..16+8h+7}.
static __device__ __forceinline__ v16h load_a_row(const _Float16* row, int h) {
    v8h lo = *(const v8h*)(row + 8 * h);
    v8h hi = *(const v8h*)(row + 16 + 8 * h);
    return __builtin_shufflevector(lo, hi, 0, 1, 2, 3, 4, 5, 6, 7,
                                           8, 9, 10, 11, 12, 13, 14, 15);
}

// out[r0+m, n0+n] = (sum_c W[r0+m, c] * x[b, c, n0+n] + bias[r0+m]) * oscale
// MODE 0: store transposed as [B][N][ROWS]  (A/B-operand friendly, packed store)
// MODE 1: store as [B][ROWS][N]
template <int MODE, int ROWS>
__global__ void __launch_bounds__(32)
proj_kernel(const float* __restrict__ x, const float* __restrict__ w,
            const float* __restrict__ bias, _Float16* __restrict__ outp,
            float oscale) {
    const int lane = threadIdx.x & 31;
    const int lr = lane & 15, h = lane >> 4;
    const int n0 = blockIdx.x * 16;
    const int r0 = blockIdx.y * 16;
    const int b  = blockIdx.z;

    v8f acc;
    #pragma unroll
    for (int r = 0; r < 8; ++r) acc[r] = 0.0f;

    const float* xb = x + (size_t)b * C_ * N_;
    #pragma unroll
    for (int c0 = 0; c0 < C_; c0 += 32) {
        // A: weight rows (M = lane%16), K chunks per ISA f16 A layout
        const float* wrow = w + (size_t)(r0 + lr) * C_ + c0;
        v16h a, bmat;
        #pragma unroll
        for (int e = 0; e < 8; ++e) {
            a[e]     = (_Float16)wrow[8 * h + e];
            a[8 + e] = (_Float16)wrow[16 + 8 * h + e];
        }
        // B: column N = n0 + lane%16, K chunks per ISA layout {8h..8h+7, 16+8h..16+8h+7}
        const float* xcol = xb + (size_t)c0 * N_ + n0 + lr;
        #pragma unroll
        for (int e = 0; e < 8; ++e) {
            bmat[e]     = (_Float16)xcol[(size_t)(8 * h + e) * N_];
            bmat[8 + e] = (_Float16)xcol[(size_t)(16 + 8 * h + e) * N_];
        }
        acc = wmma_f16(a, bmat, acc);
    }

    // D: element (M = r + 8h, N = lane%16)
    if (MODE == 0) {
        v8h pk;
        #pragma unroll
        for (int r = 0; r < 8; ++r)
            pk[r] = (_Float16)((acc[r] + bias[r0 + r + 8 * h]) * oscale);
        *(v8h*)&outp[((size_t)b * N_ + n0 + lr) * ROWS + r0 + 8 * h] = pk;
    } else {
        #pragma unroll
        for (int r = 0; r < 8; ++r) {
            const int orow = r0 + r + 8 * h;
            const float v = (acc[r] + bias[orow]) * oscale;
            outp[((size_t)b * ROWS + orow) * N_ + n0 + lr] = (_Float16)v;
        }
    }
}

#define NW 4  // waves per block; each wave owns 64 output channels

__global__ void __launch_bounds__(128)
attn_kernel(const _Float16* __restrict__ Qd,   // [B][32][N]  (pre-scaled, exp2 domain)
            const _Float16* __restrict__ Kt,   // [B][N][32]
            const _Float16* __restrict__ Vt,   // [B][N][256]
            const float* __restrict__ x,       // [B][C][N]
            float* __restrict__ out) {         // [B][C][N]
    __shared__ __attribute__((aligned(32))) float sstat[NW][16];

    const int tid = threadIdx.x;
    const int w = tid >> 5, lane = tid & 31;
    const int lr = lane & 15, h = lane >> 4;
    const int i0 = blockIdx.x * 16;
    const int b  = blockIdx.y;

    // Loop-invariant B-operand for S^T: column i = i0 + lane%16, d chunks {8h..8h+7, 16+8h..16+8h+7}
    v16h qB;
    #pragma unroll
    for (int e = 0; e < 8; ++e) {
        qB[e]     = Qd[((size_t)b * D_ + 8 * h + e) * N_ + i0 + lr];
        qB[8 + e] = Qd[((size_t)b * D_ + 16 + 8 * h + e) * N_ + i0 + lr];
    }

    // All-ones B-operand: wmma(P_a, ones_B) accumulates l_i = sum_j p_ij
    // directly in the O^T register layout (i = r + 8h) -> no transpose ever.
    v16h onesB;
    #pragma unroll
    for (int e = 0; e < 16; ++e) onesB[e] = (_Float16)1.0f;

    v8f o[4], lacc;
    #pragma unroll
    for (int t = 0; t < 4; ++t)
        #pragma unroll
        for (int r = 0; r < 8; ++r) o[t][r] = 0.0f;
    #pragma unroll
    for (int r = 0; r < 8; ++r) lacc[r] = 0.0f;

    float m_i = -1e30f;  // running max (log2 domain) for row i = lane%16

    const _Float16* kt = Kt + (size_t)b * N_ * D_;
    const _Float16* vt = Vt + (size_t)b * N_ * C_;

    for (int j0 = 0; j0 < N_; j0 += 64) {
        if (j0 + 64 < N_) {  // global_prefetch_b8 for next step's K and V rows
            __builtin_prefetch(kt + (size_t)(j0 + 64 + lr) * D_, 0, 3);
            __builtin_prefetch(kt + (size_t)(j0 + 96 + lr) * D_, 0, 3);
            __builtin_prefetch(vt + (size_t)(j0 + 64 + lane) * C_ + w * 64, 0, 3);
            __builtin_prefetch(vt + (size_t)(j0 + 96 + lane) * C_ + w * 64, 0, 3);
        }

        // ---- S^T = K Q^T over 64 keys: A = K rows (M = j local), B = Q^T (N = i = lane%16)
        const v16h ka0 = load_a_row(kt + (size_t)(j0 +      lr) * D_, h);
        const v16h ka1 = load_a_row(kt + (size_t)(j0 + 16 + lr) * D_, h);
        const v16h ka2 = load_a_row(kt + (size_t)(j0 + 32 + lr) * D_, h);
        const v16h ka3 = load_a_row(kt + (size_t)(j0 + 48 + lr) * D_, h);
        v8f s0, s1, s2, s3;
        #pragma unroll
        for (int r = 0; r < 8; ++r) { s0[r] = 0.0f; s1[r] = 0.0f; s2[r] = 0.0f; s3[r] = 0.0f; }
        s0 = wmma_f16(ka0, qB, s0);   // D[M = j local, N = i = lr]
        s1 = wmma_f16(ka1, qB, s1);
        s2 = wmma_f16(ka2, qB, s2);
        s3 = wmma_f16(ka3, qB, s3);

        // ---- online softmax: 32 scores of row i live in this lane (+ mirror half)
        float mx = fmaxf(fmaxf(s0[0], s1[0]), fmaxf(s2[0], s3[0]));
        #pragma unroll
        for (int r = 1; r < 8; ++r)
            mx = fmaxf(mx, fmaxf(fmaxf(s0[r], s1[r]), fmaxf(s2[r], s3[r])));
        mx = fmaxf(mx, __shfl_xor(mx, 16, 32));   // merge the two halves
        const float mn = fmaxf(m_i, mx);
        const float aI = __builtin_amdgcn_exp2f(m_i - mn);  // alpha, one per lane
        m_i = mn;

        // ---- P in f16: S^T D-layout IS the A-operand layout (K = j local)
        v16h pa0, pa1;
        #pragma unroll
        for (int r = 0; r < 8; ++r) {
            pa0[r]     = (_Float16)__builtin_amdgcn_exp2f(s0[r] - mn);
            pa0[8 + r] = (_Float16)__builtin_amdgcn_exp2f(s1[r] - mn);
            pa1[r]     = (_Float16)__builtin_amdgcn_exp2f(s2[r] - mn);
            pa1[8 + r] = (_Float16)__builtin_amdgcn_exp2f(s3[r] - mn);
        }

        // ---- broadcast alpha from i-layout (lane) to register layout (i = r + 8h)
        if (h == 0) sstat[w][lr] = aI;            // wave-private, in-order LDS
        const v4f al0 = *(const v4f*)&sstat[w][8 * h];
        const v4f al1 = *(const v4f*)&sstat[w][8 * h + 4];

        // ---- O^T = O^T*alpha + P @ V^T ; l = l*alpha + P @ ones  (once per 64 keys)
        #pragma unroll
        for (int t = 0; t < 4; ++t) {
            #pragma unroll
            for (int r = 0; r < 4; ++r) {
                o[t][r]     *= al0[r];
                o[t][4 + r] *= al1[r];
            }
        }
        #pragma unroll
        for (int r = 0; r < 4; ++r) { lacc[r] *= al0[r]; lacc[4 + r] *= al1[r]; }

        #pragma unroll
        for (int t = 0; t < 4; ++t) {
            // B: channel = lane%16, j local chunks {8h..8h+7, 16+8h..16+8h+7}
            const _Float16* vc = vt + (size_t)j0 * C_ + w * 64 + t * 16 + lr;
            v16h vB0, vB1;
            #pragma unroll
            for (int e = 0; e < 8; ++e) {
                vB0[e]     = vc[(size_t)(8 * h + e) * C_];
                vB0[8 + e] = vc[(size_t)(16 + 8 * h + e) * C_];
                vB1[e]     = vc[(size_t)(32 + 8 * h + e) * C_];
                vB1[8 + e] = vc[(size_t)(48 + 8 * h + e) * C_];
            }
            o[t] = wmma_f16(pa0, vB0, o[t]);      // D[M=i, N=c]
            o[t] = wmma_f16(pa1, vB1, o[t]);
        }
        lacc = wmma_f16(pa0, onesB, lacc);
        lacc = wmma_f16(pa1, onesB, lacc);
    }

    // ---- finalize: lacc[r] = l for i = r + 8h; epilogue fully vectorized
    v8f linv;
    #pragma unroll
    for (int r = 0; r < 8; ++r) linv[r] = __builtin_amdgcn_rcpf(lacc[r]);

    #pragma unroll
    for (int t = 0; t < 4; ++t) {
        const size_t row = ((size_t)b * C_ + w * 64 + t * 16 + lr) * N_ + i0 + 8 * h;
        const v4f xlo = *(const v4f*)&x[row];
        const v4f xhi = *(const v4f*)&x[row + 4];
        v4f olo, ohi;
        #pragma unroll
        for (int r = 0; r < 4; ++r) {
            olo[r] = o[t][r] * linv[r] + xlo[r];
            ohi[r] = o[t][4 + r] * linv[4 + r] + xhi[r];
        }
        *(v4f*)&out[row]     = olo;
        *(v4f*)&out[row + 4] = ohi;
    }
}

extern "C" void kernel_launch(void* const* d_in, const int* in_sizes, int n_in,
                              void* d_out, int out_size, void* d_ws, size_t ws_size,
                              hipStream_t stream) {
    const float* x  = (const float*)d_in[0];
    const float* wq = (const float*)d_in[1];
    const float* bq = (const float*)d_in[2];
    const float* wk = (const float*)d_in[3];
    const float* bk = (const float*)d_in[4];
    const float* wv = (const float*)d_in[5];
    const float* bv = (const float*)d_in[6];
    float* out = (float*)d_out;

    char* ws = (char*)d_ws;
    const size_t qk_bytes = (size_t)B_ * N_ * D_ * sizeof(_Float16);  // 2 MB each
    _Float16* Qd = (_Float16*)(ws);                    // [B][32][N], pre-scaled
    _Float16* Kt = (_Float16*)(ws + qk_bytes);         // [B][N][32]
    _Float16* Vt = (_Float16*)(ws + 2 * qk_bytes);     // [B][N][256], 16 MB

    proj_kernel<1, D_><<<dim3(N_ / 16, D_ / 16, B_), dim3(32), 0, stream>>>(x, wq, bq, Qd, QSCALE_);
    proj_kernel<0, D_><<<dim3(N_ / 16, D_ / 16, B_), dim3(32), 0, stream>>>(x, wk, bk, Kt, 1.0f);
    proj_kernel<0, C_><<<dim3(N_ / 16, C_ / 16, B_), dim3(32), 0, stream>>>(x, wv, bv, Vt, 1.0f);
    attn_kernel<<<dim3(N_ / 16, B_), dim3(NW * 32), 0, stream>>>(Qd, Kt, Vt, x, out);
}
